// NormLinearAttention_50843822850602
// MI455X (gfx1250) — matrix operand order repaired and checked by reading, not compiled, not run on hardware
//
#include <hip/hip_runtime.h>
#include <hip/hip_bf16.h>

typedef __attribute__((ext_vector_type(16))) _Float16 v16h;
typedef __attribute__((ext_vector_type(8)))  float    v8f;

#define BATCH    2
#define SEQ      2048
#define NEMBD    1024
#define NHEAD    8
#define HDIM     128      // per-head dim
#define EDIM     256      // lrpe-doubled dim
#define ROWS     (BATCH*SEQ)   // 4096

// ---------------------------------------------------------------------------
// CDNA5 async direct global->LDS copy (ASYNCcnt path), via inline asm so it
// is portable across ROCm 7.2 / amdgpu-toolchain builtin arity differences.
// LDS operand is the low 32 bits of the generic pointer (flat LDS addresses
// truncate to the in-allocation byte offset per ISA 10.2).
// ---------------------------------------------------------------------------
__device__ __forceinline__ void async_load_b128(unsigned lds_byte_addr,
                                                const void* gaddr) {
  asm volatile("global_load_async_to_lds_b128 %0, %1, off"
               :: "v"(lds_byte_addr),
                  "v"((unsigned long long)(uintptr_t)gaddr)
               : "memory");
}
__device__ __forceinline__ void wait_async0() {
  asm volatile("s_wait_asynccnt 0x0" ::: "memory");
}
__device__ __forceinline__ unsigned lds_addr(const void* p) {
  return (unsigned)(uintptr_t)p;
}

// ---------------------------------------------------------------------------
// WMMA helpers (CDNA5 wave32, 16x16x32 f16 -> f32)
// ---------------------------------------------------------------------------
__device__ __forceinline__ v8f wmma_f16(v16h a, v16h b, v8f c) {
  return __builtin_amdgcn_wmma_f32_16x16x32_f16(
      /*neg_a=*/false, a, /*neg_b=*/false, b,
      /*c_mod=*/(short)0, c, /*reuse_a=*/false, /*reuse_b=*/false);
}

// A-matrix 16x32 f16 fragment (contiguous k order, same as load_b_frag): lane holds row M=lane%16,
// VGPR v: K = 2v + 16*hi, hi=lane/16.
// Two contiguous 16B runs -> compiler coalesces to ds_load_b128 pairs.
__device__ __forceinline__ v16h load_a_frag(const _Float16* __restrict__ row, int hi) {
  v16h f;
#pragma unroll
  for (int v = 0; v < 8; ++v) {
    int kb = 2 * v + 16 * hi;
    f[2 * v]     = row[kb];
    f[2 * v + 1] = row[kb + 1];
  }
  return f;
}

// B-matrix 32x16 f16 fragment: lane holds column N=lane%16, reads along K.
// VGPR v: K = 2v + 16*hi.
__device__ __forceinline__ v16h load_b_frag(const _Float16* __restrict__ row, int hi) {
  v16h f;
#pragma unroll
  for (int v = 0; v < 8; ++v) {
    int kb = 2 * v + 16 * hi;
    f[2 * v]     = row[kb];
    f[2 * v + 1] = row[kb + 1];
  }
  return f;
}

// Same as load_b_frag but K strided (column gather from row-major V tile).
__device__ __forceinline__ v16h load_b_frag_strided(const _Float16* __restrict__ base,
                                                    int hi, int stride) {
  v16h f;
#pragma unroll
  for (int v = 0; v < 8; ++v) {
    int kb = 2 * v + 16 * hi;
    f[2 * v]     = base[kb * stride];
    f[2 * v + 1] = base[(kb + 1) * stride];
  }
  return f;
}

// ---------------------------------------------------------------------------
// Kernel 0: one-shot fp32 -> f16 conversion (amortized across GEMM tile reuse)
// ---------------------------------------------------------------------------
__global__ __launch_bounds__(256) void cvt_f32_f16(const float* __restrict__ in,
                                                   _Float16* __restrict__ out) {
  int i = blockIdx.x * 256 + threadIdx.x;
  out[i] = (_Float16)in[i];
}

// ---------------------------------------------------------------------------
// Kernel 1/5: C[M,N] = A[M,K] @ W[N,K]^T + bias  (f16 operands, f32 accum)
// 64x64 tile, BK=32, 8 waves as 4x2. Async-LDS staging: one b128 per thread
// per operand per k-step. Grid = (M/64)*(N/64).
// ---------------------------------------------------------------------------
__global__ __launch_bounds__(256) void wmma_gemm_bias(
    const _Float16* __restrict__ A, const _Float16* __restrict__ W,
    const float* __restrict__ bias, float* __restrict__ C,
    int M, int N, int K) {
  __shared__ alignas(16) _Float16 sA[64 * 32];
  __shared__ alignas(16) _Float16 sB[64 * 32];

  const int tiles_n = N >> 6;
  const int bm = blockIdx.x / tiles_n;
  const int bn = blockIdx.x % tiles_n;
  const int tid  = threadIdx.x;
  const int lane = tid & 31;
  const int wave = tid >> 5;
  const int wrow = wave >> 1;   // 0..3: 16-row block
  const int wcol = wave & 1;    // 0..1: 32-col half
  const int hi   = lane >> 4;
  const int lrow = lane & 15;

  const unsigned sAb = lds_addr(sA);
  const unsigned sBb = lds_addr(sB);
  const int cprow = tid >> 2;            // 0..63  (tile row this thread copies)
  const int cpcol = (tid & 3) * 8;       // 0/8/16/24 (half offset)

  v8f acc0 = {};
  v8f acc1 = {};

  for (int k0 = 0; k0 < K; k0 += 32) {
    async_load_b128(sAb + tid * 16,
                    A + (size_t)(bm * 64 + cprow) * K + k0 + cpcol);
    async_load_b128(sBb + tid * 16,
                    W + (size_t)(bn * 64 + cprow) * K + k0 + cpcol);
    wait_async0();
    __syncthreads();

    v16h a  = load_a_frag(&sA[(wrow * 16 + lrow) * 32], hi);
    v16h b0 = load_b_frag(&sB[(wcol * 32 + lrow) * 32], hi);
    v16h b1 = load_b_frag(&sB[(wcol * 32 + 16 + lrow) * 32], hi);
    acc0 = wmma_f16(a, b0, acc0);
    acc1 = wmma_f16(a, b1, acc1);
    __syncthreads();
  }

  const int gn0 = bn * 64 + wcol * 32 + lrow;
  const int gn1 = gn0 + 16;
  const float bv0 = bias ? bias[gn0] : 0.0f;
  const float bv1 = bias ? bias[gn1] : 0.0f;
#pragma unroll
  for (int v = 0; v < 8; ++v) {
    int gm = bm * 64 + wrow * 16 + v + 8 * hi;   // C layout: M = v + 8*hi
    C[(size_t)gm * N + gn0] = acc0[v] + bv0;
    C[(size_t)gm * N + gn1] = acc1[v] + bv1;
  }
}

// ---------------------------------------------------------------------------
// Kernel 2: silu + lrpe; qkvu fp32 row -> Qh/Kh (f16, [b,h,n,256]) + Vh (f16)
// ---------------------------------------------------------------------------
__global__ __launch_bounds__(256) void prep_kernel(
    const float* __restrict__ qkvu, const float* __restrict__ theta,
    _Float16* __restrict__ Qh, _Float16* __restrict__ Kh,
    _Float16* __restrict__ Vh) {
  const int row = blockIdx.x;            // b*SEQ + m
  const int b = row >> 11;
  const int m = row & (SEQ - 1);
  const int tid = threadIdx.x;
  const float* r = qkvu + (size_t)row * (4 * NEMBD);

#pragma unroll
  for (int i = 0; i < 4; ++i) {
    int c = tid + i * 256;               // 0..1023 == h*128 + d
    int h = c >> 7, d = c & 127;
    float th = theta[c] * (float)m;
    float sth, cth;
    sincosf(th, &sth, &cth);

    size_t qbase = ((size_t)(b * NHEAD + h) * SEQ + m) * EDIM;
    size_t vbase = ((size_t)(b * NHEAD + h) * SEQ + m) * HDIM;

    float qv = r[c];
    qv = qv / (1.0f + expf(-qv));        // silu
    Qh[qbase + d]        = (_Float16)(qv * cth);
    Qh[qbase + HDIM + d] = (_Float16)(qv * sth);

    float kv = r[NEMBD + c];
    kv = kv / (1.0f + expf(-kv));
    Kh[qbase + d]        = (_Float16)(kv * cth);
    Kh[qbase + HDIM + d] = (_Float16)(kv * sth);

    Vh[vbase + d] = (_Float16)r[2 * NEMBD + c];
  }
}

// ---------------------------------------------------------------------------
// Kernel 3: attention. One WG per (bh, 64-row tile). S = Q K^T (*mask), O=S V.
// mask[i][j] = exp(slope) for j<=i else 1 (pure elementwise scale, no softmax).
// 8 waves: 4 row-blocks x 2 column halves of the 64x128 O tile.
// ---------------------------------------------------------------------------
__global__ __launch_bounds__(256) void attn_kernel(
    const _Float16* __restrict__ Qh, const _Float16* __restrict__ Kh,
    const _Float16* __restrict__ Vh, const float* __restrict__ slope,
    float* __restrict__ O) {
  __shared__ alignas(16) _Float16 sQ[64 * EDIM];     // 32 KB
  __shared__ alignas(16) _Float16 sK[32 * EDIM];     // 16 KB
  __shared__ alignas(16) _Float16 sV[32 * HDIM];     //  8 KB
  __shared__ alignas(16) _Float16 sS[8][16 * 32];    //  8 KB per-wave S bounce

  const int bh = blockIdx.y;             // 0..15
  const int it = blockIdx.x;             // 0..31 (64-row tile)
  const float expS = expf(slope[bh & (NHEAD - 1)]);

  const int tid  = threadIdx.x;
  const int lane = tid & 31;
  const int wave = tid >> 5;
  const int wrow = wave >> 1;
  const int wcol = wave & 1;
  const int hi   = lane >> 4;
  const int lrow = lane & 15;

  // Stage the 64x256 Q tile once (32 KB = 8 async b128 per thread).
  {
    const char* gq = (const char*)(Qh + ((size_t)bh * SEQ + it * 64) * EDIM);
    const unsigned sqb = lds_addr(sQ);
#pragma unroll
    for (int i = 0; i < 8; ++i) {
      unsigned off = (unsigned)(tid + i * 256) * 16;
      async_load_b128(sqb + off, gq + off);
    }
    wait_async0();
  }
  __syncthreads();

  v16h qf[8];
#pragma unroll
  for (int ks = 0; ks < 8; ++ks)
    qf[ks] = load_a_frag(&sQ[(wrow * 16 + lrow) * EDIM + ks * 32], hi);

  v8f acc[4] = {};
  const unsigned skb = lds_addr(sK);
  const unsigned svb = lds_addr(sV);

  for (int j0 = 0; j0 < SEQ; j0 += 32) {
    __syncthreads();   // previous iteration's sK/sV reads complete
    {
      const char* gk = (const char*)(Kh + ((size_t)bh * SEQ + j0) * EDIM);
#pragma unroll
      for (int i = 0; i < 4; ++i) {
        unsigned off = (unsigned)(tid + i * 256) * 16;
        async_load_b128(skb + off, gk + off);
      }
      const char* gv = (const char*)(Vh + ((size_t)bh * SEQ + j0) * HDIM);
#pragma unroll
      for (int i = 0; i < 2; ++i) {
        unsigned off = (unsigned)(tid + i * 256) * 16;
        async_load_b128(svb + off, gv + off);
      }
      wait_async0();
    }
    __syncthreads();

    // S = Q K^T for this wave's 16 rows x 32 j-columns, masked, -> f16 in sS.
#pragma unroll
    for (int nb = 0; nb < 2; ++nb) {
      v8f s = {};
#pragma unroll
      for (int ks = 0; ks < 8; ++ks) {
        v16h bf = load_b_frag(&sK[(nb * 16 + lrow) * EDIM + ks * 32], hi);
        s = wmma_f16(qf[ks], bf, s);
      }
      const int ibase = it * 64 + wrow * 16;
      const int j = j0 + nb * 16 + lrow;         // C layout: N = lane%16
#pragma unroll
      for (int v = 0; v < 8; ++v) {
        int i = ibase + v + 8 * hi;              // C layout: M = v + 8*hi
        float val = s[v] * ((j <= i) ? expS : 1.0f);
        sS[wave][(v + 8 * hi) * 32 + nb * 16 + lrow] = (_Float16)val;
      }
    }
    // Same-wave LDS write->read; order DS ops explicitly.
    asm volatile("s_wait_dscnt 0x0" ::: "memory");

    v16h a2 = load_a_frag(&sS[wave][lrow * 32], hi);
#pragma unroll
    for (int cb = 0; cb < 4; ++cb) {
      v16h bv = load_b_frag_strided(&sV[wcol * 64 + cb * 16 + lrow], hi, HDIM);
      acc[cb] = wmma_f16(a2, bv, acc[cb]);
    }
  }

#pragma unroll
  for (int cb = 0; cb < 4; ++cb) {
    int col = wcol * 64 + cb * 16 + lrow;
#pragma unroll
    for (int v = 0; v < 8; ++v) {
      int i = it * 64 + wrow * 16 + v + 8 * hi;
      O[((size_t)bh * SEQ + i) * HDIM + col] = acc[cb][v];
    }
  }
}

// ---------------------------------------------------------------------------
// Kernel 4: y[b,m,:] = srmsnorm(O gathered head-major) * u   (emits f16 y)
// ---------------------------------------------------------------------------
__global__ __launch_bounds__(256) void norm_mul_kernel(
    const float* __restrict__ O, const float* __restrict__ qkvu,
    _Float16* __restrict__ y) {
  __shared__ float sred[256];
  const int row = blockIdx.x;            // b*SEQ + m
  const int b = row >> 11;
  const int m = row & (SEQ - 1);
  const int tid = threadIdx.x;

  float vals[4];
  float ss = 0.0f;
#pragma unroll
  for (int i = 0; i < 4; ++i) {
    int c = tid + i * 256;               // h*128 + d
    int h = c >> 7, d = c & 127;
    float o = O[(((size_t)(b * NHEAD + h)) * SEQ + m) * HDIM + d];
    vals[i] = o;
    ss += o * o;
  }
  sred[tid] = ss;
  __syncthreads();
  for (int s = 128; s > 0; s >>= 1) {
    if (tid < s) sred[tid] += sred[tid + s];
    __syncthreads();
  }
  float scale = rsqrtf(sred[0] * (1.0f / (float)NEMBD) + 1e-6f);
#pragma unroll
  for (int i = 0; i < 4; ++i) {
    int c = tid + i * 256;
    float u = qkvu[(size_t)row * (4 * NEMBD) + 3 * NEMBD + c];
    y[(size_t)row * NEMBD + c] = (_Float16)(vals[i] * scale * u);
  }
}

// ---------------------------------------------------------------------------
extern "C" void kernel_launch(void* const* d_in, const int* in_sizes, int n_in,
                              void* d_out, int out_size, void* d_ws, size_t ws_size,
                              hipStream_t stream) {
  const float* x       = (const float*)d_in[0];
  const float* slope   = (const float*)d_in[1];
  const float* qkvu_w  = (const float*)d_in[2];
  const float* qkvu_b  = (const float*)d_in[3];
  const float* out_w   = (const float*)d_in[4];
  const float* out_b   = (const float*)d_in[5];
  const float* theta   = (const float*)d_in[6];
  float* out = (float*)d_out;

  char* ws = (char*)d_ws;
  const size_t MB = (size_t)1 << 20;
  float*    qkvu = (float*)(ws);                  // 4096x4096 f32   = 64 MB
  _Float16* Qh   = (_Float16*)(ws + 64 * MB);     // [2,8,2048,256]  = 16 MB
  _Float16* Kh   = (_Float16*)(ws + 80 * MB);     // 16 MB
  _Float16* Vh   = (_Float16*)(ws + 96 * MB);     // [2,8,2048,128]  =  8 MB
  float*    Obuf = (float*)(ws + 104 * MB);       // f32             = 16 MB
  _Float16* yh   = (_Float16*)(ws + 120 * MB);    // [4096,1024] f16 =  8 MB
  _Float16* xh   = (_Float16*)(ws + 128 * MB);    // x f16           =  8 MB
  _Float16* wqh  = (_Float16*)(ws + 136 * MB);    // qkvu_w f16      =  8 MB
  _Float16* woh  = (_Float16*)(ws + 144 * MB);    // out_w f16       =  2 MB
  (void)in_sizes; (void)n_in; (void)out_size; (void)ws_size;

  // 0) one-shot f16 conversions (amortized over GEMM tile reuse)
  cvt_f32_f16<<<dim3((ROWS * NEMBD) / 256), 256, 0, stream>>>(x, xh);
  cvt_f32_f16<<<dim3((4 * NEMBD * NEMBD) / 256), 256, 0, stream>>>(qkvu_w, wqh);
  cvt_f32_f16<<<dim3((NEMBD * NEMBD) / 256), 256, 0, stream>>>(out_w, woh);

  // 1) qkvu = x @ qkvu_w^T + qkvu_b   (M=4096, N=4096, K=1024)
  wmma_gemm_bias<<<dim3((ROWS / 64) * (4 * NEMBD / 64)), 256, 0, stream>>>(
      xh, wqh, qkvu_b, qkvu, ROWS, 4 * NEMBD, NEMBD);

  // 2) silu + lrpe -> f16 Q/K/V in head-major layout
  prep_kernel<<<dim3(ROWS), 256, 0, stream>>>(qkvu, theta, Qh, Kh, Vh);

  // 3) attention: grid (row tiles, b*h)
  attn_kernel<<<dim3(SEQ / 64, BATCH * NHEAD), 256, 0, stream>>>(
      Qh, Kh, Vh, slope, Obuf);

  // 4) srmsnorm + u-gate -> f16 y
  norm_mul_kernel<<<dim3(ROWS), 256, 0, stream>>>(Obuf, qkvu, yh);

  // 5) out = y @ out_w^T + out_b   (M=4096, N=1024, K=1024)
  wmma_gemm_bias<<<dim3((ROWS / 64) * (NEMBD / 64)), 256, 0, stream>>>(
      yh, woh, out_b, out, ROWS, NEMBD, NEMBD);
}
